// MSHGNN_13632226198283
// MI455X (gfx1250) — matrix operand order repaired and checked by reading, not compiled, not run on hardware
//
#include <hip/hip_runtime.h>
#include <hip/hip_bf16.h>

// Problem constants (match reference)
#define N_NODES 50000
#define E_EDGES 100000
#define G_GRAPHS 5000
#define DIN  128
#define HEADS 8
#define DOUT 128
#define HD   (HEADS*DOUT)   // 1024

typedef float  v2f  __attribute__((ext_vector_type(2)));
typedef float  v8f  __attribute__((ext_vector_type(8)));
typedef __bf16 v8bf __attribute__((ext_vector_type(8)));
typedef __bf16 v16bf __attribute__((ext_vector_type(16)));

__device__ __forceinline__ void split_bf16(float x, __bf16* hi, __bf16* lo) {
    __bf16 h = (__bf16)x;
    *hi = h;
    *lo = (__bf16)(x - (float)h);
}

// ---------------------------------------------------------------------------
// Prep (one kernel, three index regions):
//  R1: W_fc  -> transposed bf16 hi/lo      [4][1024 cols][128 K]
//  R2: sum_r W_res[r] -> transposed hi/lo  [1024][128]   (residuals collapse)
//  R3: ALR[i, r*16+side*8+h] = sum_d W_fc[r][i][h*128+d]*attn[r][h][d]
//      -> transposed hi/lo                 [64][128]
// ---------------------------------------------------------------------------
__global__ void mshgnn_prep(const float* __restrict__ W_fc,
                            const float* __restrict__ attn_l,
                            const float* __restrict__ attn_r,
                            const float* __restrict__ W_res,
                            __bf16* __restrict__ WfcT_hi, __bf16* __restrict__ WfcT_lo,
                            __bf16* __restrict__ WresT_hi, __bf16* __restrict__ WresT_lo,
                            __bf16* __restrict__ ALRT_hi, __bf16* __restrict__ ALRT_lo)
{
    int tid = blockIdx.x * blockDim.x + threadIdx.x;

    if (tid < 4 * DIN * HD) {                     // R1: transpose-convert W_fc
        int r   = tid / (DIN * HD);
        int rem = tid - r * DIN * HD;
        int i   = rem / HD;                        // input row (K)
        int c   = rem - i * HD;                    // output col
        float x = W_fc[tid];
        size_t di = (size_t)r * HD * DIN + (size_t)c * DIN + i;
        split_bf16(x, &WfcT_hi[di], &WfcT_lo[di]);
        return;
    }
    int t2 = tid - 4 * DIN * HD;
    if (t2 < DIN * HD) {                           // R2: WresSum transposed
        int i = t2 / HD, c = t2 - i * HD;
        float s = 0.f;
        #pragma unroll
        for (int r = 0; r < 4; r++) s += W_res[(size_t)r * DIN * HD + (size_t)i * HD + c];
        size_t di = (size_t)c * DIN + i;
        split_bf16(s, &WresT_hi[di], &WresT_lo[di]);
        return;
    }
    int t3 = t2 - DIN * HD;
    if (t3 < DIN * 64) {                           // R3: attention logit matrix
        int i = t3 >> 6;                           // input row 0..127
        int c = t3 & 63;                           // packed col
        int r = c >> 4;
        int side = (c >> 3) & 1;                   // 0 = attn_l, 1 = attn_r
        int h = c & 7;
        const float* av = (side == 0 ? attn_l : attn_r) + r * HEADS * DOUT + h * DOUT;
        const float* wv = W_fc + (size_t)r * DIN * HD + (size_t)i * HD + h * DOUT;
        float s = 0.f;
        for (int d = 0; d < DOUT; d++) s += wv[d] * av[d];
        size_t di = (size_t)c * DIN + i;
        split_bf16(s, &ALRT_hi[di], &ALRT_lo[di]);
    }
}

// ---------------------------------------------------------------------------
// Split feat into bf16 hi/lo pair (row-major [N,128]).
// ---------------------------------------------------------------------------
__global__ void mshgnn_split_feat(const float* __restrict__ feat,
                                  __bf16* __restrict__ fhi, __bf16* __restrict__ flo)
{
    int idx = blockIdx.x * blockDim.x + threadIdx.x;
    if (idx >= N_NODES * DIN) return;
    split_bf16(feat[idx], &fhi[idx], &flo[idx]);
}

// ---------------------------------------------------------------------------
// bf16x3 split GEMM: C[M,Ncols] = (Ahi+Alo) @ (Bhi+Blo), fp32 accumulate via
// V_WMMA_F32_16X16X32_BF16, dropping the Alo*Blo term (<= 2^-16 rel).
// One wave -> 16 rows x 64 cols (4 accumulators). K = 128 -> 4 k-steps,
// 4 tiles x 3 products = 48 WMMAs/wave.
// A fragment (16x32, 16-bit, ISA 7.12.2): lane half=lane>>4;
//   elems 0-7 = K k0+half*8.., elems 8-15 = K k0+16+half*8..
// B fragment (32x16): column = lane&15, K same as A (k0+half*8.., k0+16+half*8..)
//   -> B stored TRANSPOSED [col][K] so the 16 K-values are contiguous.
// ---------------------------------------------------------------------------
__global__ void mshgnn_gemm_bf16x3(const __bf16* __restrict__ Ahi,
                                   const __bf16* __restrict__ Alo,
                                   const __bf16* __restrict__ BThi,
                                   const __bf16* __restrict__ BTlo,
                                   float* __restrict__ C,
                                   int M, int Ncols)
{
    int lane = threadIdx.x & 31;
    int wid  = (blockIdx.x * blockDim.x + threadIdx.x) >> 5;
    int tilesN = Ncols >> 6;
    int tm = wid / tilesN;
    int tn = wid - tm * tilesN;
    if (tm * 16 >= M) return;                     // whole-wave uniform exit

    int row0 = tm << 4, col0 = tn << 6;
    int half = lane >> 4;                         // 0 or 1
    int aRow = row0 + (lane & 15);
    const __bf16* ah = Ahi + (size_t)aRow * DIN + half * 8;
    const __bf16* al = Alo + (size_t)aRow * DIN + half * 8;
    // per-tile B base: column (col0 + j*16 + lane&15), K offset half*8
    size_t bBase0 = (size_t)(col0 +  0 + (lane & 15)) * DIN + half * 8;
    size_t bBase1 = (size_t)(col0 + 16 + (lane & 15)) * DIN + half * 8;
    size_t bBase2 = (size_t)(col0 + 32 + (lane & 15)) * DIN + half * 8;
    size_t bBase3 = (size_t)(col0 + 48 + (lane & 15)) * DIN + half * 8;

    v8f c0 = {}, c1 = {}, c2 = {}, c3 = {};

    #pragma unroll
    for (int ks = 0; ks < 4; ks++) {
        int k0 = ks * 32;
        __builtin_prefetch(ah + k0 + 32, 0, 0);   // next A hi chunk

        v8bf h0 = *(const v8bf*)(ah + k0);
        v8bf h1 = *(const v8bf*)(ah + k0 + 16);
        v8bf l0 = *(const v8bf*)(al + k0);
        v8bf l1 = *(const v8bf*)(al + k0 + 16);
        v16bf a_hi = __builtin_shufflevector(h0, h1, 0,1,2,3,4,5,6,7,8,9,10,11,12,13,14,15);
        v16bf a_lo = __builtin_shufflevector(l0, l1, 0,1,2,3,4,5,6,7,8,9,10,11,12,13,14,15);

        #define DO_TILE(cacc, bBase)                                                        \
        {                                                                                   \
            v8bf bh0 = *(const v8bf*)(BThi + (bBase) + k0);                                 \
            v8bf bh1 = *(const v8bf*)(BThi + (bBase) + k0 + 16);                            \
            v8bf bl0 = *(const v8bf*)(BTlo + (bBase) + k0);                                 \
            v8bf bl1 = *(const v8bf*)(BTlo + (bBase) + k0 + 16);                            \
            v16bf b_hi = __builtin_shufflevector(bh0, bh1,                                  \
                             0,1,2,3,4,5,6,7,8,9,10,11,12,13,14,15);                        \
            v16bf b_lo = __builtin_shufflevector(bl0, bl1,                                  \
                             0,1,2,3,4,5,6,7,8,9,10,11,12,13,14,15);                        \
            cacc = __builtin_amdgcn_wmma_f32_16x16x32_bf16(false, a_lo, false, b_hi,        \
                                                           (short)0, cacc, false, false);   \
            cacc = __builtin_amdgcn_wmma_f32_16x16x32_bf16(false, a_hi, false, b_lo,        \
                                                           (short)0, cacc, false, false);   \
            cacc = __builtin_amdgcn_wmma_f32_16x16x32_bf16(false, a_hi, false, b_hi,        \
                                                           (short)0, cacc, false, false);   \
        }
        DO_TILE(c0, bBase0)
        DO_TILE(c1, bBase1)
        DO_TILE(c2, bBase2)
        DO_TILE(c3, bBase3)
        #undef DO_TILE
    }

    // C/D layout: VGPR i -> row0 + i (lanes 0-15) / row0 + 8 + i (lanes 16-31)
    int crow = row0 + (half << 3);
    float* Cp = C + (size_t)crow * Ncols + col0 + (lane & 15);
    #pragma unroll
    for (int i = 0; i < 8; i++) {
        Cp[(size_t)i * Ncols +  0] = c0[i];
        Cp[(size_t)i * Ncols + 16] = c1[i];
        Cp[(size_t)i * Ncols + 32] = c2[i];
        Cp[(size_t)i * Ncols + 48] = c3[i];
    }
}

// ---------------------------------------------------------------------------
// Edge logits + softmax denominators for one relation.
// ---------------------------------------------------------------------------
__global__ void mshgnn_edge_logits(const int* __restrict__ src,
                                   const int* __restrict__ dst,
                                   const float* __restrict__ ELR, int rel,
                                   float* __restrict__ pbuf,
                                   float* __restrict__ denom)
{
    int idx = blockIdx.x * blockDim.x + threadIdx.x;
    if (idx >= E_EDGES * HEADS) return;
    int e = idx >> 3, h = idx & 7;
    int s = src[e], d = dst[e];
    float x = ELR[s * 64 + rel * 16 + h] + ELR[d * 64 + rel * 16 + 8 + h];
    x = x > 0.f ? x : 0.2f * x;                    // leaky_relu(0.2)
    float p = __expf(x);                           // shift-free softmax: logits O(1)
    pbuf[idx] = p;
    atomicAdd(&denom[d * HEADS + h], p);
}

// ---------------------------------------------------------------------------
// Scatter: rstAcc[dst, h, :] += Z[src, h, :] * (p / denom[dst, h]).
// ---------------------------------------------------------------------------
__global__ void mshgnn_edge_scatter(const int* __restrict__ src,
                                    const int* __restrict__ dst,
                                    const float* __restrict__ Z,
                                    const float* __restrict__ pbuf,
                                    const float* __restrict__ denom,
                                    float* __restrict__ rstAcc)
{
    int e = blockIdx.x;
    int s = src[e], d = dst[e];
    const float* zs = Z + (size_t)s * HD;
    float* rd = rstAcc + (size_t)d * HD;
    for (int j = threadIdx.x; j < HD; j += blockDim.x) {
        int h = j >> 7;
        float a = pbuf[e * HEADS + h] / denom[d * HEADS + h];
        atomicAdd(&rd[j], zs[j] * a);
    }
}

// ---------------------------------------------------------------------------
// Per-graph mean accumulation of input features.
// ---------------------------------------------------------------------------
__global__ void mshgnn_graph_accum(const float* __restrict__ feat,
                                   const int* __restrict__ gid,
                                   float* __restrict__ gsums,
                                   float* __restrict__ gcnt)
{
    int idx = blockIdx.x * blockDim.x + threadIdx.x;
    if (idx >= N_NODES * DIN) return;
    int n = idx >> 7, d = idx & 127;
    int g = gid[n];
    atomicAdd(&gsums[g * DIN + d], feat[idx]);
    if (d == 0) atomicAdd(&gcnt[g], 1.f);
}

// ---------------------------------------------------------------------------
// Finalize: out[n,d] = gmean[gid[n],d] + max_h rstAcc[n,h,d]
// ---------------------------------------------------------------------------
__global__ void mshgnn_finalize(const float* __restrict__ rstAcc,
                                const int* __restrict__ gid,
                                const float* __restrict__ gsums,
                                const float* __restrict__ gcnt,
                                float* __restrict__ out)
{
    int idx = blockIdx.x * blockDim.x + threadIdx.x;
    if (idx >= N_NODES * DOUT) return;
    int n = idx >> 7, d = idx & 127;
    const float* rp = rstAcc + (size_t)n * HD + d;
    float m = rp[0];
    #pragma unroll
    for (int h = 1; h < HEADS; h++) m = fmaxf(m, rp[h * DOUT]);
    int g = gid[n];
    float c = gcnt[g]; c = c > 1.f ? c : 1.f;
    out[idx] = gsums[g * DIN + d] / c + m;
}

// ---------------------------------------------------------------------------
extern "C" void kernel_launch(void* const* d_in, const int* in_sizes, int n_in,
                              void* d_out, int out_size, void* d_ws, size_t ws_size,
                              hipStream_t stream) {
    const float* feat      = (const float*)d_in[0];
    const float* W_fc      = (const float*)d_in[1];
    const float* attn_l    = (const float*)d_in[2];
    const float* attn_r    = (const float*)d_in[3];
    const float* W_res     = (const float*)d_in[4];
    const int*   src_intra = (const int*)d_in[5];
    const int*   dst_intra = (const int*)d_in[6];
    const int*   src_inter = (const int*)d_in[7];
    const int*   dst_inter = (const int*)d_in[8];
    const int*   graph_id  = (const int*)d_in[9];
    float* out = (float*)d_out;

    // Workspace carving (~460 MB total)
    char* ws = (char*)d_ws;
    size_t off = 0;
    auto carve = [&](size_t bytes) -> void* {
        void* p = ws + off;
        off = (off + bytes + 255) & ~(size_t)255;
        return p;
    };
    float*  rstAcc  = (float*)carve((size_t)N_NODES * HD * 4);
    float*  Z       = (float*)carve((size_t)N_NODES * HD * 4);
    float*  ELR     = (float*)carve((size_t)N_NODES * 64 * 4);
    float*  pbuf    = (float*)carve((size_t)E_EDGES * HEADS * 4);
    float*  denom   = (float*)carve((size_t)N_NODES * HEADS * 4);
    float*  gsums   = (float*)carve((size_t)G_GRAPHS * DIN * 4);
    float*  gcnt    = (float*)carve((size_t)G_GRAPHS * 4);
    __bf16* fhi     = (__bf16*)carve((size_t)N_NODES * DIN * 2);
    __bf16* flo     = (__bf16*)carve((size_t)N_NODES * DIN * 2);
    __bf16* WfcT_hi = (__bf16*)carve((size_t)4 * HD * DIN * 2);
    __bf16* WfcT_lo = (__bf16*)carve((size_t)4 * HD * DIN * 2);
    __bf16* WresT_hi= (__bf16*)carve((size_t)HD * DIN * 2);
    __bf16* WresT_lo= (__bf16*)carve((size_t)HD * DIN * 2);
    __bf16* ALRT_hi = (__bf16*)carve((size_t)64 * DIN * 2);
    __bf16* ALRT_lo = (__bf16*)carve((size_t)64 * DIN * 2);

    // 1) Prep: weight transforms + bf16 hi/lo splits
    {
        int total = 4 * DIN * HD + DIN * HD + DIN * 64;
        mshgnn_prep<<<(total + 255) / 256, 256, 0, stream>>>(
            W_fc, attn_l, attn_r, W_res,
            WfcT_hi, WfcT_lo, WresT_hi, WresT_lo, ALRT_hi, ALRT_lo);
    }
    mshgnn_split_feat<<<(N_NODES * DIN + 255) / 256, 256, 0, stream>>>(feat, fhi, flo);

    // 2) Per-graph feature means (independent of GAT path)
    hipMemsetAsync(gsums, 0, (size_t)G_GRAPHS * DIN * 4, stream);
    hipMemsetAsync(gcnt,  0, (size_t)G_GRAPHS * 4, stream);
    mshgnn_graph_accum<<<(N_NODES * DIN + 255) / 256, 256, 0, stream>>>(feat, graph_id,
                                                                        gsums, gcnt);

    // 3) rstAcc = feat @ (sum_r W_res[r])  — initializes the [N,H,D] accumulator
    int bigWaves = (N_NODES / 16) * (HD / 64);     // 50000 waves
    mshgnn_gemm_bf16x3<<<(bigWaves * 32 + 255) / 256, 256, 0, stream>>>(
        fhi, flo, WresT_hi, WresT_lo, rstAcc, N_NODES, HD);

    // 4) ELR = feat @ ALR  — all el/er logits for 4 relations in one GEMM
    {
        int waves = (N_NODES / 16);                // Ncols = 64 -> 1 tile group
        mshgnn_gemm_bf16x3<<<(waves * 32 + 255) / 256, 256, 0, stream>>>(
            fhi, flo, ALRT_hi, ALRT_lo, ELR, N_NODES, 64);
    }

    // 5) Four relations: r0 intra, r1 inter (conv1), r2/r3 reversed (conv2)
    const int* srcs[4] = { src_intra, src_inter, dst_intra, dst_inter };
    const int* dsts[4] = { dst_intra, dst_inter, src_intra, src_inter };
    for (int r = 0; r < 4; r++) {
        mshgnn_gemm_bf16x3<<<(bigWaves * 32 + 255) / 256, 256, 0, stream>>>(
            fhi, flo, WfcT_hi + (size_t)r * HD * DIN, WfcT_lo + (size_t)r * HD * DIN,
            Z, N_NODES, HD);
        hipMemsetAsync(denom, 0, (size_t)N_NODES * HEADS * 4, stream);
        mshgnn_edge_logits<<<(E_EDGES * HEADS + 255) / 256, 256, 0, stream>>>(
            srcs[r], dsts[r], ELR, r, pbuf, denom);
        mshgnn_edge_scatter<<<E_EDGES, 256, 0, stream>>>(
            srcs[r], dsts[r], Z, pbuf, denom, rstAcc);
    }

    // 6) out = graph_mean[gid] + max_heads(rstAcc)
    mshgnn_finalize<<<(N_NODES * DOUT + 255) / 256, 256, 0, stream>>>(
        rstAcc, graph_id, gsums, gcnt, out);
}
